// SyntheticTransformer_35364760715638
// MI455X (gfx1250) — matrix operand order repaired and checked by reading, not compiled, not run on hardware
//
#include <hip/hip_runtime.h>
#include <hip/hip_bf16.h>
#include <stdint.h>

typedef __attribute__((ext_vector_type(16))) _Float16 v16h;
typedef __attribute__((ext_vector_type(8)))  _Float16 v8h;
typedef __attribute__((ext_vector_type(8)))  float    v8f;
typedef __attribute__((ext_vector_type(4)))  unsigned int u32x4;
typedef __attribute__((ext_vector_type(8)))  int      i32x8;
typedef __attribute__((ext_vector_type(4)))  int      i32x4;

#define L_SEQ  2048
#define DMODEL 512
#define NHEAD  8
#define NLAYER 4
#define DHEAD  64
#define DROT   32
#define DQK    96
#define DFFN   2048

// Use the CDNA5 Tensor Data Mover for the GEMM A-tile (f16 raw 2D tile -> LDS).
#define USE_TDM 1

__device__ __forceinline__ v8f wmma_f16(v16h a, v16h b, v8f c) {
  // D = A(16x32 f16) x B(32x16 f16) + C(16x16 f32)
  return __builtin_amdgcn_wmma_f32_16x16x32_f16(false, a, false, b, (short)0, c,
                                                false, false);
}

__device__ __forceinline__ v16h combine16(v8h lo, v8h hi) {
  v16h r;
#pragma unroll
  for (int j = 0; j < 8; ++j) { r[j] = lo[j]; r[8 + j] = hi[j]; }
  return r;
}

__device__ __forceinline__ v8f zero8f() {
  v8f z = {0.f, 0.f, 0.f, 0.f, 0.f, 0.f, 0.f, 0.f};
  return z;
}

// ---------------------------------------------------------------- gather
__global__ __launch_bounds__(256) void gather_kernel(const int* __restrict__ ids,
                                                     const float* __restrict__ emb,
                                                     float* __restrict__ x) {
  int i = blockIdx.x * 256 + threadIdx.x;
  if (i < L_SEQ * DMODEL) {
    int l = i / DMODEL, d = i - l * DMODEL;
    x[i] = emb[(size_t)ids[l] * DMODEL + d];
  }
}

// ---------------------------------------------------------------- rmsnorm -> f16
__global__ __launch_bounds__(256) void rmsnorm_kernel(const float* __restrict__ x,
                                                      const float* __restrict__ w,
                                                      _Float16* __restrict__ out) {
  int row = blockIdx.x;
  const float* xr = x + (size_t)row * DMODEL;
  float s = 0.f;
  for (int j = threadIdx.x; j < DMODEL; j += 256) { float v = xr[j]; s += v * v; }
  __shared__ float red[8];
#pragma unroll
  for (int m = 16; m >= 1; m >>= 1) s += __shfl_xor(s, m, 32);
  if ((threadIdx.x & 31) == 0) red[threadIdx.x >> 5] = s;
  __syncthreads();
  if (threadIdx.x == 0) {
    float t = 0.f;
#pragma unroll
    for (int i = 0; i < 8; ++i) t += red[i];
    red[0] = rsqrtf(t / (float)DMODEL + 1e-6f);
  }
  __syncthreads();
  float inv = red[0];
  for (int j = threadIdx.x; j < DMODEL; j += 256)
    out[(size_t)row * DMODEL + j] = (_Float16)(xr[j] * inv * w[j]);
}

// ---------------------------------------------------------------- PoPE
__device__ __forceinline__ float softplusf(float x) {
  return (x > 20.f) ? x : log1pf(expf(x));
}

// one wave per (l, h): builds q2/k2 [H][L][96] f16 and v-transposed [H][64][L] f16
__global__ __launch_bounds__(128) void pope_kernel(const float* __restrict__ q,
                                                   const float* __restrict__ k,
                                                   const float* __restrict__ v,
                                                   const float* __restrict__ qr_w,
                                                   const float* __restrict__ kr_w,
                                                   const float* __restrict__ delta,
                                                   _Float16* __restrict__ q2,
                                                   _Float16* __restrict__ k2,
                                                   _Float16* __restrict__ vT) {
  int wave = blockIdx.x * 4 + (threadIdx.x >> 5);
  int lane = threadIdx.x & 31;
  int h = wave & (NHEAD - 1);
  int l = wave >> 3;
  const float* qrow = q + (size_t)l * DMODEL + h * DHEAD;
  const float* krow = k + (size_t)l * DMODEL + h * DHEAD;
  const float* vrow = v + (size_t)l * DMODEL + h * DHEAD;

  float qv = qrow[lane], kv = krow[lane];
  float sq = qv * qv, sk = kv * kv;
#pragma unroll
  for (int m = 1; m <= 16; m <<= 1) { sq += __shfl_xor(sq, m, 32); sk += __shfl_xor(sk, m, 32); }
  float qn = qv * rsqrtf(sq / (float)DROT + 1e-6f) * qr_w[lane];
  float kn = kv * rsqrtf(sk / (float)DROT + 1e-6f) * kr_w[lane];
  float mu_q = softplusf(qn), mu_k = softplusf(kn);

  float theta = expf(-((float)lane / (float)DROT) * 9.210340371976184f); // ln(1e4)
  float ph = (float)l * theta;
  float dl = delta[lane];
  dl = fminf(0.f, fmaxf(dl, -6.283185307179586f));
  float phk = ph + dl;

  size_t qb = ((size_t)h * L_SEQ + l) * DQK;
  q2[qb + lane]          = (_Float16)qrow[DROT + lane];
  q2[qb + DROT + lane]   = (_Float16)(mu_q * cosf(ph));
  q2[qb + 2*DROT + lane] = (_Float16)(mu_q * sinf(ph));
  k2[qb + lane]          = (_Float16)krow[DROT + lane];
  k2[qb + DROT + lane]   = (_Float16)(mu_k * cosf(phk));
  k2[qb + 2*DROT + lane] = (_Float16)(mu_k * sinf(phk));

  vT[((size_t)h * DHEAD + lane) * L_SEQ + l]        = (_Float16)vrow[lane];
  vT[((size_t)h * DHEAD + 32 + lane) * L_SEQ + l]   = (_Float16)vrow[32 + lane];
}

// ---------------------------------------------------------------- flash attention
// one wave per (head, 16-query-row tile); streams 32-key chunks
__global__ __launch_bounds__(128) void flash_kernel(const _Float16* __restrict__ q2,
                                                    const _Float16* __restrict__ k2,
                                                    const _Float16* __restrict__ vT,
                                                    _Float16* __restrict__ o16) {
  __shared__ __align__(16) _Float16 pbuf[4][16 * 32];
  int wid = threadIdx.x >> 5, lane = threadIdx.x & 31;
  int gw = blockIdx.x * 4 + wid;
  int h = gw & (NHEAD - 1);
  int qt = gw >> 3;
  int q0 = qt * 16;
  int half = lane >> 4, ln = lane & 15;
  const float NEG_INF = -__builtin_inff();

  // Q fragments (A layout): 3 chunks of K=32 over the 96-dim feature axis
  v16h qa[3];
#pragma unroll
  for (int c = 0; c < 3; ++c) {
    const _Float16* p = q2 + ((size_t)h * L_SEQ + q0 + ln) * DQK + c * 32 + half * 8;
    qa[c] = combine16(*(const v8h*)p, *(const v8h*)(p + 16));
  }

  float mrow[8], lrow[8];
  v8f oacc[4];
#pragma unroll
  for (int v = 0; v < 8; ++v) { mrow[v] = NEG_INF; lrow[v] = 0.f; }
#pragma unroll
  for (int jc = 0; jc < 4; ++jc) oacc[jc] = zero8f();

  int ktmax = (q0 + 15) >> 5;
  for (int kt = 0; kt <= ktmax; ++kt) {
    int k0 = kt * 32;
    // S = Q x K^T for 32 keys (two 16-key sub-tiles)
    v8f s[2] = {zero8f(), zero8f()};
#pragma unroll
    for (int nt = 0; nt < 2; ++nt) {
      int key = k0 + nt * 16 + ln;
#pragma unroll
      for (int c = 0; c < 3; ++c) {
        const _Float16* p = k2 + ((size_t)h * L_SEQ + key) * DQK + c * 32 + half * 8;
        v16h bf = combine16(*(const v8h*)p, *(const v8h*)(p + 16));
        s[nt] = wmma_f16(qa[c], bf, s[nt]);
      }
    }
    // online softmax over this 32-key chunk
    float p0[8], p1[8], corr[8];
#pragma unroll
    for (int v = 0; v < 8; ++v) {
      int qrow = q0 + half * 8 + v;
      float s0 = (k0 + ln > qrow)      ? NEG_INF : s[0][v] * 0.125f;
      float s1 = (k0 + 16 + ln > qrow) ? NEG_INF : s[1][v] * 0.125f;
      float mx = fmaxf(s0, s1);
#pragma unroll
      for (int m = 1; m <= 8; m <<= 1) mx = fmaxf(mx, __shfl_xor(mx, m, 32));
      mx = fmaxf(mx, mrow[v]);
      float e0 = __expf(s0 - mx), e1 = __expf(s1 - mx);
      float ps = e0 + e1;
#pragma unroll
      for (int m = 1; m <= 8; m <<= 1) ps += __shfl_xor(ps, m, 32);
      float c = __expf(mrow[v] - mx);
      lrow[v] = lrow[v] * c + ps;
      mrow[v] = mx;
      corr[v] = c; p0[v] = e0; p1[v] = e1;
    }
#pragma unroll
    for (int jc = 0; jc < 4; ++jc)
#pragma unroll
      for (int v = 0; v < 8; ++v) oacc[jc][v] *= corr[v];

    // re-layout P (C layout -> A layout) through per-wave LDS tile
    _Float16* pb = &pbuf[wid][0];
#pragma unroll
    for (int v = 0; v < 8; ++v) {
      int row = half * 8 + v;
      pb[row * 32 + ln]      = (_Float16)p0[v];
      pb[row * 32 + 16 + ln] = (_Float16)p1[v];
    }
    // same-wave DS ops are in order: stores above complete before loads below
    v16h pa = combine16(*(const v8h*)(pb + ln * 32 + half * 8),
                        *(const v8h*)(pb + ln * 32 + half * 8 + 16));
    // O += P x V  (V pre-transposed: contiguous along keys)
#pragma unroll
    for (int jc = 0; jc < 4; ++jc) {
      const _Float16* vb = vT + ((size_t)h * DHEAD + jc * 16 + ln) * L_SEQ + k0 + half * 8;
      v16h bf = combine16(*(const v8h*)vb, *(const v8h*)(vb + 16));
      oacc[jc] = wmma_f16(pa, bf, oacc[jc]);
    }
  }
  // normalize + store f16 output [L, D] with col = h*64 + d
#pragma unroll
  for (int jc = 0; jc < 4; ++jc)
#pragma unroll
    for (int v = 0; v < 8; ++v) {
      int row = q0 + half * 8 + v;
      int col = h * DHEAD + jc * 16 + ln;
      o16[(size_t)row * DMODEL + col] = (_Float16)(oacc[jc][v] / lrow[v]);
    }
}

// ---------------------------------------------------------------- WMMA GEMM
// C[M,N] = A_f16[M,K] x B_f32[K,N] (B converted to f16 on the fly).
// MODE 0: store f32   MODE 1: C = Res + acc (f32)   MODE 2: store f16(gelu(acc))
// TRANSB: B is [N,K] row-major (for logits = h @ emb^T)
// A tile (64x32 f16, raw copy) is staged into LDS by the Tensor Data Mover:
// wave 0 issues tensor_load_to_lds with a 2D D# (tile 32x64 elems, stride K),
// waits on TENSORcnt, then the block barrier publishes the tile.
template <int MODE, bool TRANSB>
__global__ __launch_bounds__(256) void gemm_kernel(const _Float16* __restrict__ A,
                                                   const float* __restrict__ B,
                                                   const float* __restrict__ Res,
                                                   float* __restrict__ Cf,
                                                   _Float16* __restrict__ Ch,
                                                   int M, int N, int K) {
  // NOTE: As is the first __shared__ object -> LDS offset 0 (used as D#.lds_addr)
  __shared__ __align__(16) _Float16 As[64 * 32];   // [m][k]
  __shared__ __align__(16) _Float16 Bs[128 * 32];  // [n][k] (k-transposed)
  int tid = threadIdx.x, lane = tid & 31, w = tid >> 5;
  int half = lane >> 4, ln = lane & 15;
  int wm = w & 1, wn = w >> 1;                      // 2 x 4 wave grid
  int bm0 = blockIdx.y * 64, bn0 = blockIdx.x * 128;

  v8f acc[2][2];
#pragma unroll
  for (int i = 0; i < 2; ++i)
#pragma unroll
    for (int j = 0; j < 2; ++j) acc[i][j] = zero8f();

  for (int kt = 0; kt < K; kt += 32) {
#if USE_TDM
    if (w == 0) {  // TDM: DMA the 64x32 f16 A tile into LDS (offset 0)
      uint64_t ga = (uint64_t)(uintptr_t)(A + (size_t)bm0 * K + kt);
      unsigned td0 = (unsigned)K, td1 = (unsigned)M;
      uint64_t st0 = (uint64_t)(unsigned)K;
      u32x4 g0;
      g0[0] = 1u;                                             // count=1, user mode
      g0[1] = 0u;                                             // lds_addr = &As = 0
      g0[2] = (unsigned)ga;                                   // global_addr[31:0]
      g0[3] = (unsigned)((ga >> 32) & 0x1FFFFFFu) | (2u << 30); // addr[56:32]|type=2
      i32x8 g1;
      g1[0] = (int)(1u << 16);                                // data_size=1 (2 bytes)
      g1[1] = (int)((td0 & 0xFFFFu) << 16);                   // tensor_dim0[15:0]
      g1[2] = (int)((td0 >> 16) | ((td1 & 0xFFFFu) << 16));   // dim0 hi | dim1 lo
      g1[3] = (int)((td1 >> 16) | (32u << 16));               // dim1 hi | tile_dim0=32
      g1[4] = (int)64u;                                       // tile_dim1=64, tile_dim2=0
      g1[5] = (int)(unsigned)st0;                             // dim0_stride[31:0]
      g1[6] = (int)((unsigned)(st0 >> 32) & 0xFFFFu);         // dim0_stride[47:32]
      g1[7] = 0;                                              // dim1_stride (unused)
      i32x4 g2 = {0, 0, 0, 0}, g3 = {0, 0, 0, 0};
      asm volatile("tensor_load_to_lds %0, %1, %2, %3"
                   :: "s"(g0), "s"(g1), "s"(g2), "s"(g3)
                   : "memory");
      asm volatile("s_wait_tensorcnt 0x0" ::: "memory");
    }
#else
    {  // stage A: 64x32 halves, one b128 per thread
      int row = tid >> 2, kc = (tid & 3) * 8;
      *(v8h*)(As + row * 32 + kc) =
          *(const v8h*)(A + (size_t)(bm0 + row) * K + kt + kc);
    }
#endif
    if (!TRANSB) {  // stage B [K,N] -> Bs[n][k], f32 -> f16 on the fly
#pragma unroll
      for (int t = 0; t < 2; ++t) {
        int idx = tid + t * 256;
        int kr = idx >> 4, nc = (idx & 15) * 8;
        const float* src = B + (size_t)(kt + kr) * N + bn0 + nc;
#pragma unroll
        for (int j = 0; j < 8; ++j) Bs[(nc + j) * 32 + kr] = (_Float16)src[j];
        if (kt + 32 < K) __builtin_prefetch(src + (size_t)32 * N, 0, 1);
      }
    } else {  // stage B [N,K] -> Bs[n][k]
#pragma unroll
      for (int t = 0; t < 2; ++t) {
        int idx = tid + t * 256;
        int n = idx >> 2, kc = (idx & 3) * 8;
        const float* src = B + (size_t)(bn0 + n) * K + kt + kc;
#pragma unroll
        for (int j = 0; j < 8; ++j) Bs[n * 32 + kc + j] = (_Float16)src[j];
        if (kt + 32 < K) __builtin_prefetch(src + 32, 0, 1);
      }
    }
    __syncthreads();

    v16h af[2], bf[2];
#pragma unroll
    for (int i = 0; i < 2; ++i) {
      int r = wm * 32 + i * 16 + ln;
      af[i] = combine16(*(const v8h*)(As + r * 32 + half * 8),
                        *(const v8h*)(As + r * 32 + half * 8 + 16));
    }
#pragma unroll
    for (int j = 0; j < 2; ++j) {
      int n = wn * 32 + j * 16 + ln;
      bf[j] = combine16(*(const v8h*)(Bs + n * 32 + half * 8),
                        *(const v8h*)(Bs + n * 32 + half * 8 + 16));
    }
#pragma unroll
    for (int i = 0; i < 2; ++i)
#pragma unroll
      for (int j = 0; j < 2; ++j) acc[i][j] = wmma_f16(af[i], bf[j], acc[i][j]);
    __syncthreads();
  }

#pragma unroll
  for (int i = 0; i < 2; ++i)
#pragma unroll
    for (int j = 0; j < 2; ++j)
#pragma unroll
      for (int v = 0; v < 8; ++v) {
        int row = bm0 + wm * 32 + i * 16 + half * 8 + v;
        int col = bn0 + wn * 32 + j * 16 + ln;
        size_t off = (size_t)row * N + col;
        float val = acc[i][j][v];
        if (MODE == 0) {
          Cf[off] = val;
        } else if (MODE == 1) {
          Cf[off] = Res[off] + val;
        } else {  // exact GELU, f16 output
          float g = 0.5f * val * (1.f + erff(val * 0.7071067811865475f));
          Ch[off] = (_Float16)g;
        }
      }
}

// ---------------------------------------------------------------- launch
extern "C" void kernel_launch(void* const* d_in, const int* in_sizes, int n_in,
                              void* d_out, int out_size, void* d_ws, size_t ws_size,
                              hipStream_t stream) {
  (void)in_sizes; (void)n_in; (void)out_size; (void)ws_size;
  const int*   ids = (const int*)d_in[0];
  const float* emb = (const float*)d_in[1];
  const float* wq  = (const float*)d_in[2];
  const float* wk  = (const float*)d_in[3];
  const float* wv  = (const float*)d_in[4];
  const float* wo  = (const float*)d_in[5];
  const float* n1  = (const float*)d_in[6];
  const float* n2  = (const float*)d_in[7];
  const float* qrw = (const float*)d_in[8];
  const float* krw = (const float*)d_in[9];
  const float* dlt = (const float*)d_in[10];
  const float* w1  = (const float*)d_in[11];
  const float* w2  = (const float*)d_in[12];
  const float* fnw = (const float*)d_in[13];
  float* out = (float*)d_out;

  char* p = (char*)d_ws;
  auto alloc = [&](size_t bytes) {
    char* r = p;
    p += (bytes + 255) & ~(size_t)255;
    return r;
  };
  float*    x   = (float*)alloc(sizeof(float) * L_SEQ * DMODEL);
  float*    qf  = (float*)alloc(sizeof(float) * L_SEQ * DMODEL);
  float*    kf  = (float*)alloc(sizeof(float) * L_SEQ * DMODEL);
  float*    vf  = (float*)alloc(sizeof(float) * L_SEQ * DMODEL);
  _Float16* h16 = (_Float16*)alloc(2 * (size_t)L_SEQ * DMODEL);
  _Float16* g16 = (_Float16*)alloc(2 * (size_t)L_SEQ * DFFN);
  _Float16* o16 = (_Float16*)alloc(2 * (size_t)L_SEQ * DMODEL);
  _Float16* q2h = (_Float16*)alloc(2 * (size_t)NHEAD * L_SEQ * DQK);
  _Float16* k2h = (_Float16*)alloc(2 * (size_t)NHEAD * L_SEQ * DQK);
  _Float16* vTh = (_Float16*)alloc(2 * (size_t)NHEAD * DHEAD * L_SEQ);

  dim3 blk(256);
  dim3 g512(DMODEL / 128, L_SEQ / 64);
  dim3 gffn(DFFN / 128, L_SEQ / 64);
  dim3 glog(32000 / 128, L_SEQ / 64);

  gather_kernel<<<(L_SEQ * DMODEL + 255) / 256, blk, 0, stream>>>(ids, emb, x);

  for (int i = 0; i < NLAYER; ++i) {
    rmsnorm_kernel<<<L_SEQ, blk, 0, stream>>>(x, n1 + (size_t)i * DMODEL, h16);
    gemm_kernel<0, false><<<g512, blk, 0, stream>>>(
        h16, wq + (size_t)i * DMODEL * DMODEL, nullptr, qf, nullptr, L_SEQ, DMODEL, DMODEL);
    gemm_kernel<0, false><<<g512, blk, 0, stream>>>(
        h16, wk + (size_t)i * DMODEL * DMODEL, nullptr, kf, nullptr, L_SEQ, DMODEL, DMODEL);
    gemm_kernel<0, false><<<g512, blk, 0, stream>>>(
        h16, wv + (size_t)i * DMODEL * DMODEL, nullptr, vf, nullptr, L_SEQ, DMODEL, DMODEL);
    pope_kernel<<<L_SEQ * NHEAD / 4, 128, 0, stream>>>(
        qf, kf, vf, qrw + (size_t)i * DROT, krw + (size_t)i * DROT,
        dlt + (size_t)i * DROT, q2h, k2h, vTh);
    flash_kernel<<<NHEAD * (L_SEQ / 16) / 4, 128, 0, stream>>>(q2h, k2h, vTh, o16);
    gemm_kernel<1, false><<<g512, blk, 0, stream>>>(
        o16, wo + (size_t)i * DMODEL * DMODEL, x, x, nullptr, L_SEQ, DMODEL, DMODEL);
    rmsnorm_kernel<<<L_SEQ, blk, 0, stream>>>(x, n2 + (size_t)i * DMODEL, h16);
    gemm_kernel<2, false><<<gffn, blk, 0, stream>>>(
        h16, w1 + (size_t)i * DMODEL * DFFN, nullptr, nullptr, g16, L_SEQ, DFFN, DMODEL);
    gemm_kernel<1, false><<<g512, blk, 0, stream>>>(
        g16, w2 + (size_t)i * DFFN * DMODEL, x, x, nullptr, L_SEQ, DMODEL, DFFN);
  }

  rmsnorm_kernel<<<L_SEQ, blk, 0, stream>>>(x, fnw, h16);
  gemm_kernel<0, true><<<glog, blk, 0, stream>>>(
      h16, emb, nullptr, out, nullptr, L_SEQ, 32000, DMODEL);
}
